// LSTM_4209067950580
// MI455X (gfx1250) — 8-bit matrix operand order repaired and checked by reading, compiled, not run on hardware
//
#include <hip/hip_runtime.h>
#include <hip/hip_bf16.h>

typedef __attribute__((ext_vector_type(16))) int   v16i;
typedef __attribute__((ext_vector_type(8)))  float v8f;

#define B_    8192
#define T_    2048
#define H_    256
#define C_    10
#define BTILE 64
#define SROW  272

__device__ __forceinline__ unsigned f32_to_fp8_manual(float x) {
  union { float f; unsigned u; } v; v.f = x;
  unsigned s = (v.u >> 24) & 0x80u;
  float a = __builtin_fabsf(x);
  if (a != a || a >= 448.0f) return s | 0x7Eu;
  if (a < 0x1.0p-10f) return s;
  if (a < 0x1.0p-6f) {
    int q = (int)(a * 512.0f + 0.5f);
    if (q > 7) return s | 0x08u;
    return s | (unsigned)q;
  }
  unsigned u = v.u;
  unsigned exp = (u >> 23) & 0xFFu;
  unsigned man = u & 0x7FFFFFu;
  unsigned keep = man >> 20;
  unsigned lsb = keep & 1u;
  unsigned rb = (man >> 19) & 1u;
  unsigned st = (man & 0x7FFFFu) ? 1u : 0u;
  keep += (rb & (st | lsb));
  if (keep > 7u) { keep = 0u; exp += 1u; }
  int e8 = (int)exp - 127 + 7;
  if (e8 >= 16) return s | 0x7Eu;
  if (e8 <= 0)  return s;
  return s | ((unsigned)e8 << 3) | keep;
}

__device__ __forceinline__ unsigned f32_to_fp8(float x) {
#if __has_builtin(__builtin_amdgcn_cvt_pk_fp8_f32)
  return (unsigned)(__builtin_amdgcn_cvt_pk_fp8_f32(x, x, 0, false)) & 0xFFu;
#else
  return f32_to_fp8_manual(x);
#endif
}

__device__ __forceinline__ float fp8_to_f32(int b) {
  int e = (b >> 3) & 0xF;
  int m = b & 7;
  float mag = (e == 0) ? ldexpf((float)m, -9) : ldexpf((float)(8 + m), e - 10);
  return (b & 0x80) ? -mag : mag;
}

__device__ __forceinline__ float tanh_fast(float x) {
#if __has_builtin(__builtin_amdgcn_tanhf)
  return __builtin_amdgcn_tanhf(x);
#else
  return tanhf(x);
#endif
}
__device__ __forceinline__ float sigm(float x) {
  return 0.5f + 0.5f * tanh_fast(0.5f * x);
}

__global__ __launch_bounds__(256, 1)
void lstm_fp8_wmma_kernel(const float* __restrict__ x,
                          const float* __restrict__ Wgx, const float* __restrict__ Wgh,
                          const float* __restrict__ Wix, const float* __restrict__ Wih,
                          const float* __restrict__ Wfx, const float* __restrict__ Wfh,
                          const float* __restrict__ Wox, const float* __restrict__ Woh,
                          const float* __restrict__ Wph,
                          const float* __restrict__ bg,  const float* __restrict__ bi,
                          const float* __restrict__ bf,  const float* __restrict__ bo,
                          const float* __restrict__ bp,
                          float* __restrict__ out)
{
  __shared__ char  hbuf[2][BTILE * SROW];
  __shared__ float xbuf[2][BTILE];

  const int tid  = threadIdx.x;
  const int wave = tid >> 5;
  const int lane = tid & 31;
  const int l16  = lane & 15;
  const int hi   = lane >> 4;
  const int row0 = blockIdx.x * BTILE;

  for (int i = tid; i < (BTILE * SROW) / 4; i += 256)
    ((int*)hbuf[0])[i] = 0;
  if (tid < BTILE)
    xbuf[0][tid] = x[(size_t)(row0 + tid) * T_ + 0];

  const float* WxP[4] = {Wgx, Wix, Wfx, Wox};
  const float* WhP[4] = {Wgh, Wih, Wfh, Woh};
  const float* bP [4] = {bg,  bi,  bf,  bo };
  float wgx_r[4][2], bia_r[4][2];
#pragma unroll
  for (int g = 0; g < 4; ++g)
#pragma unroll
    for (int ct = 0; ct < 2; ++ct) {
      const int col = wave * 32 + ct * 16 + l16;
      wgx_r[g][ct] = WxP[g][col];
      bia_r[g][ct] = bP[g][col];
    }

  v16i Bf[4][2][2];
#pragma unroll
  for (int g = 0; g < 4; ++g) {
    const float* W = WhP[g];
#pragma unroll
    for (int ct = 0; ct < 2; ++ct) {
      const int n = wave * 32 + ct * 16 + l16;
#pragma unroll
      for (int kf = 0; kf < 2; ++kf) {
#pragma unroll
        for (int v = 0; v < 16; ++v) {
          const int Kb = kf * 128 + (v >> 2) * 32 + hi * 16 + (v & 3) * 4;
          unsigned w = 0;
#pragma unroll
          for (int b = 0; b < 4; ++b)
            w |= f32_to_fp8(W[(size_t)(Kb + b) * H_ + n]) << (8 * b);
          Bf[g][ct][kf][v] = (int)w;
        }
      }
    }
  }

  const v8f vzero = {0.f, 0.f, 0.f, 0.f, 0.f, 0.f, 0.f, 0.f};
  v8f cst[4][2];
#pragma unroll
  for (int rt = 0; rt < 4; ++rt)
#pragma unroll
    for (int ct = 0; ct < 2; ++ct) cst[rt][ct] = vzero;

  for (int t = 0; t < T_; ++t) {
    __syncthreads();

    if (tid < BTILE && (t + 1) < T_)
      xbuf[(t + 1) & 1][tid] = x[(size_t)(row0 + tid) * T_ + (t + 1)];

    const int rd = t & 1, wr = rd ^ 1;
    const int*   hread  = (const int*)hbuf[rd];
    char*        hwrite = hbuf[wr];
    const float* xcur   = xbuf[rd];

#pragma unroll
    for (int rt = 0; rt < 4; ++rt) {
      float xr[8];
#pragma unroll
      for (int j = 0; j < 8; ++j)
        xr[j] = xcur[rt * 16 + hi * 8 + j];

      v16i a0, a1;
      const int rowoff = (rt * 16 + l16) * SROW;
#pragma unroll
      for (int v = 0; v < 16; ++v) {
        const int K = (v >> 2) * 32 + hi * 16 + (v & 3) * 4;
        a0[v] = hread[(rowoff + K) >> 2];
        a1[v] = hread[(rowoff + 128 + K) >> 2];
      }

#pragma unroll
      for (int ct = 0; ct < 2; ++ct) {
        v8f acc[4];
#pragma unroll
        for (int g = 0; g < 4; ++g) {
          v8f z = vzero;
          z = __builtin_amdgcn_wmma_f32_16x16x128_fp8_fp8(a0, Bf[g][ct][0],
                                                          (short)0, z, false, false);
          z = __builtin_amdgcn_wmma_f32_16x16x128_fp8_fp8(a1, Bf[g][ct][1],
                                                          (short)0, z, false, false);
          acc[g] = z;
        }

        const int col = wave * 32 + ct * 16 + l16;
#pragma unroll
        for (int j = 0; j < 8; ++j) {
          const float xv = xr[j];
          const float pg = acc[0][j] + xv * wgx_r[0][ct] + bia_r[0][ct];
          const float pi = acc[1][j] + xv * wgx_r[1][ct] + bia_r[1][ct];
          const float pf = acc[2][j] + xv * wgx_r[2][ct] + bia_r[2][ct];
          const float po = acc[3][j] + xv * wgx_r[3][ct] + bia_r[3][ct];
          const float gv = tanh_fast(pg);
          const float iv = sigm(pi);
          const float fv = sigm(pf);
          const float ov = sigm(po);
          const float cv = gv * iv + cst[rt][ct][j] * fv;
          cst[rt][ct][j] = cv;
          const float hv = tanh_fast(cv) * ov;
          const int rrow = rt * 16 + hi * 8 + j;
          hwrite[rrow * SROW + col] = (char)f32_to_fp8(hv);
        }
      }
    }
  }

  __syncthreads();

  const unsigned char* hfin = (const unsigned char*)hbuf[T_ & 1];
  for (int idx = tid; idx < BTILE * C_; idx += 256) {
    const int r = idx / C_, cls = idx % C_;
    float acc = bp[cls];
    for (int k = 0; k < H_; ++k)
      acc += fp8_to_f32(hfin[r * SROW + k]) * Wph[k * C_ + cls];
    out[(size_t)(row0 + r) * C_ + cls] = acc;
  }
}

extern "C" void kernel_launch(void* const* d_in, const int* in_sizes, int n_in,
                              void* d_out, int out_size, void* d_ws, size_t ws_size,
                              hipStream_t stream) {
  (void)in_sizes; (void)n_in; (void)out_size; (void)d_ws; (void)ws_size;
  lstm_fp8_wmma_kernel<<<B_ / BTILE, 256, 0, stream>>>(
      (const float*)d_in[0],
      (const float*)d_in[1],  (const float*)d_in[2],
      (const float*)d_in[3],  (const float*)d_in[4],
      (const float*)d_in[5],  (const float*)d_in[6],
      (const float*)d_in[7],  (const float*)d_in[8],
      (const float*)d_in[9],
      (const float*)d_in[10], (const float*)d_in[11],
      (const float*)d_in[12], (const float*)d_in[13],
      (const float*)d_in[14],
      (float*)d_out);
}
